// RobonetConvLSTM_24043226923853
// MI455X (gfx1250) — matrix operand order repaired and checked by reading, not compiled, not run on hardware
//
#include <hip/hip_runtime.h>
#include <hip/hip_bf16.h>

// ---------------------------------------------------------------------------
// RobonetConvLSTM for MI455X (gfx1250, wave32, WMMA)
// T=40, B=256, C=32, H=W=8, D=2048. Convs run as tap-decomposed implicit GEMMs
// through v_wmma_f32_16x16x32_f16 (f16 A/B, f32 accumulate). Attention is an
// L2-resident f16-buffer two-pass kernel (bandwidth bound, no matrix math).
// Tap loops are nested constant-bound ky/kx loops so all LDS / weight offsets
// fold to immediates (no runtime div/mod in the hot loop).
// ---------------------------------------------------------------------------

typedef __attribute__((ext_vector_type(16))) _Float16 v16h;
typedef __attribute__((ext_vector_type(8)))  _Float16 v8h;
typedef __attribute__((ext_vector_type(8)))  float    v8f;

#define TT 40
#define BB 256
#define DD 2048   // 32ch * 64px

__device__ __forceinline__ float sigm(float x) {
    return 1.0f / (1.0f + __expf(-x));
}

// ------------------------- init: zero LSTM states --------------------------
__global__ void zero_states_kernel(float* p, int n) {
    int i = blockIdx.x * blockDim.x + threadIdx.x;
    for (; i < n; i += gridDim.x * blockDim.x) p[i] = 0.0f;
}

// --------------- weight prep: f32 (oc,ic,ky,kx) -> f16 [tap][oc][ic] -------
__global__ void prep_weights_kernel(const float* __restrict__ W0,
                                    const float* __restrict__ W1,
                                    _Float16* __restrict__ w0h,   // [25][128][64]
                                    _Float16* __restrict__ w1h) { // [9][128][64]
    const int n0 = 128 * 64 * 25;
    const int n1 = 128 * 64 * 9;
    int i = blockIdx.x * blockDim.x + threadIdx.x;
    if (i < n0) {
        int tap = i % 25;
        int ic  = (i / 25) % 64;
        int oc  = i / (25 * 64);
        w0h[(tap * 128 + oc) * 64 + ic] = (_Float16)W0[i];
    } else if (i < n0 + n1) {
        int j = i - n0;
        int tap = j % 9;
        int ic  = (j / 9) % 64;
        int oc  = j / (9 * 64);
        w1h[(tap * 128 + oc) * 64 + ic] = (_Float16)W1[j];
    }
}

// ------------------------- ConvLSTM cell (implicit GEMM) -------------------
// One workgroup per batch image b. 8 waves. GEMM: M=128 (out chans),
// N=64 (pixels), K = KS*KS taps x 64 in-channels, decomposed per tap so the
// K slices are contiguous channel runs in the (y,x,ch) LDS tile.
template <int KS, int PAD, bool IS_CELL1>
__global__ __launch_bounds__(256)
void convlstm_cell_kernel(const float* __restrict__ xin,  // [B][2048] (x_t or att)
                          float* h,                        // [B][2048] state (r/w)
                          float* c,                        // [B][2048] state (r/w)
                          const _Float16* __restrict__ wt, // [KS*KS][128][64] f16
                          const float* __restrict__ bias,  // [128]
                          _Float16* buf16_t,               // cell0: buf + t*B*D
                          float* out_t) {                  // cell1: out + t*B*D
    constexpr int LW = 8 + 2 * PAD;               // padded tile width
    __shared__ __align__(32) _Float16 inp[LW * LW * 64];
    __shared__ __align__(32) float gates[128 * 64];

    const int b   = blockIdx.x;
    const int tid = threadIdx.x;

    // ---- zero padded tile, then stage concat(xin, h) as (y,x,ch) f16 ----
    for (int i = tid; i < (LW * LW * 64) / 2; i += 256)
        ((unsigned int*)inp)[i] = 0u;
    __syncthreads();
    for (int i = tid; i < 4096; i += 256) {
        int ch = i >> 6, px = i & 63;
        int y = px >> 3, x = px & 7;
        float v = (ch < 32) ? xin[(size_t)b * DD + ch * 64 + px]
                            : h[(size_t)b * DD + (ch - 32) * 64 + px];
        inp[((y + PAD) * LW + (x + PAD)) * 64 + ch] = (_Float16)v;
    }
    __syncthreads();

    // ---- tap-decomposed GEMM via v_wmma_f32_16x16x32_f16 ----
    const int wave  = tid >> 5;
    const int lane  = tid & 31;
    const int l15   = lane & 15;
    const int lhalf = lane >> 4;      // 0: low K half, 1: high K half
    const int m0    = wave * 16;      // this wave's 16 output channels

    v8f acc[4] = {v8f{}, v8f{}, v8f{}, v8f{}};

    // Per-lane invariant bases; all tap/nb offsets below are compile-time.
    const _Float16* wbase = wt + (size_t)(m0 + l15) * 64 + lhalf * 16;
    const int py = l15 >> 3;          // pixel-row part within an N block (p = nb*16+l15)
    const int px0 = l15 & 7;
    // inp base for nb block: pixels p = nb*16 + l15 -> y = 2*nb + py, x = px0
    const _Float16* ibase = &inp[(py * LW + px0) * 64 + lhalf * 16];

    for (int ky = 0; ky < KS; ++ky) {
        #pragma unroll
        for (int kx = 0; kx < KS; ++kx) {
            const int tap = ky * KS + kx;
            #pragma unroll
            for (int c0 = 0; c0 < 64; c0 += 32) {
                // A fragment: 16 oc rows x 32 K (contiguous K order: k = 16*lhalf + s)
                const _Float16* ap = wbase + (size_t)tap * 128 * 64 + c0;
                v8h lo = *(const v8h*)ap;         // K = 16*lhalf+0..7
                v8h hi = *(const v8h*)(ap + 8);   // K = 16*lhalf+8..15
                v16h a = __builtin_shufflevector(lo, hi,
                         0, 1, 2, 3, 4, 5, 6, 7, 8, 9, 10, 11, 12, 13, 14, 15);
                #pragma unroll
                for (int nb = 0; nb < 4; ++nb) {
                    // y = 2*nb + py + ky, x = px0 + kx (all-constant tap shift)
                    const v16h bfr = *(const v16h*)
                        (ibase + ((2 * nb + ky) * LW + kx) * 64 + c0);
                    acc[nb] = __builtin_amdgcn_wmma_f32_16x16x32_f16(
                        false, a, false, bfr, (short)0, acc[nb], false, false);
                }
            }
        }
    }

    // ---- spill gates (+bias) to LDS in [oc][pixel] f32 ----
    const float bv0 = bias[m0 + lhalf * 8 + 0];
    #pragma unroll
    for (int r = 0; r < 8; ++r) {
        const int oc = m0 + r + lhalf * 8;    // C/D layout: lanes>=16 hold M=r+8
        const float bv = (r == 0) ? bv0 : bias[oc];
        #pragma unroll
        for (int nb = 0; nb < 4; ++nb)
            gates[oc * 64 + nb * 16 + l15] = acc[nb][r] + bv;
    }
    __syncthreads();

    // ---- fused LSTM pointwise: i,f,o,g -> new c,h ----
    for (int i = tid; i < DD; i += 256) {
        const int ch = i >> 6, px = i & 63;
        const float ig = gates[ch * 64 + px];
        const float fg = gates[(32 + ch) * 64 + px];
        const float og = gates[(64 + ch) * 64 + px];
        const float cg = gates[(96 + ch) * 64 + px];
        const size_t gi = (size_t)b * DD + i;
        const float cn = sigm(fg) * c[gi] + sigm(ig) * tanhf(cg);
        const float hn = sigm(og) * tanhf(cn);
        c[gi] = cn;
        h[gi] = hn;
        if constexpr (!IS_CELL1) buf16_t[gi] = (_Float16)hn;  // feed attention
        else                     out_t[gi]  = hn;             // final output
    }
}

// ---------------------- attention over previous encodings ------------------
// One WG per batch b. dots[j] = <h0_t, h0_j> over D=2048 (f16 buffer, f32
// accum), softmax over j<=t, att = sum_j w_j * h0_j. Buffer is L2-resident.
__global__ __launch_bounds__(256)
void attention_kernel(const _Float16* __restrict__ buf16,  // [T][B][D] f16
                      float* __restrict__ att,             // [B][D] f32
                      int t) {
    __shared__ float dots[64];
    __shared__ float wts[64];
    const int b   = blockIdx.x;
    const int tid = threadIdx.x;
    const int d0  = tid * 8;   // 2048 / 256

    if (tid < 64) dots[tid] = 0.0f;
    __syncthreads();

    float my[8];
    {
        const _Float16* fl = buf16 + ((size_t)t * BB + b) * DD + d0;
        #pragma unroll
        for (int i = 0; i < 8; ++i) my[i] = (float)fl[i];
    }

    for (int j = 0; j <= t; ++j) {
        const _Float16* row = buf16 + ((size_t)j * BB + b) * DD + d0;
        float p = 0.0f;
        #pragma unroll
        for (int i = 0; i < 8; ++i) p += my[i] * (float)row[i];
        #pragma unroll
        for (int m = 16; m >= 1; m >>= 1) p += __shfl_xor(p, m, 32);
        if ((tid & 31) == 0) atomicAdd(&dots[j], p);
    }
    __syncthreads();

    if (tid == 0) {  // tiny (<=41 entries) softmax
        float mx = -1e30f;
        for (int j = 0; j <= t; ++j) mx = fmaxf(mx, dots[j]);
        float s = 0.0f;
        for (int j = 0; j <= t; ++j) { float e = __expf(dots[j] - mx); wts[j] = e; s += e; }
        const float inv = 1.0f / s;
        for (int j = 0; j <= t; ++j) wts[j] *= inv;
    }
    __syncthreads();

    float a[8] = {0, 0, 0, 0, 0, 0, 0, 0};
    for (int j = 0; j <= t; ++j) {
        const float w = wts[j];
        const _Float16* row = buf16 + ((size_t)j * BB + b) * DD + d0;
        #pragma unroll
        for (int i = 0; i < 8; ++i) a[i] += w * (float)row[i];
    }
    #pragma unroll
    for (int i = 0; i < 8; ++i) att[(size_t)b * DD + d0 + i] = a[i];
}

// ---------------------------------------------------------------------------
extern "C" void kernel_launch(void* const* d_in, const int* in_sizes, int n_in,
                              void* d_out, int out_size, void* d_ws, size_t ws_size,
                              hipStream_t stream) {
    const float* x  = (const float*)d_in[0];   // [T][B][32][8][8]
    const float* W0 = (const float*)d_in[1];   // [128][64][5][5]
    const float* b0 = (const float*)d_in[2];   // [128]
    const float* W1 = (const float*)d_in[3];   // [128][64][3][3]
    const float* b1 = (const float*)d_in[4];   // [128]
    float* out = (float*)d_out;                // [T][B][32][8][8]

    // workspace carve-up (~50.5 MB)
    char* ws = (char*)d_ws;
    size_t off = 0;
    float* h0 = (float*)(ws + off); off += (size_t)BB * DD * 4;
    float* c0 = (float*)(ws + off); off += (size_t)BB * DD * 4;
    float* h1 = (float*)(ws + off); off += (size_t)BB * DD * 4;
    float* c1 = (float*)(ws + off); off += (size_t)BB * DD * 4;
    float* att = (float*)(ws + off); off += (size_t)BB * DD * 4;
    _Float16* buf16 = (_Float16*)(ws + off); off += (size_t)TT * BB * DD * 2;
    _Float16* w0h   = (_Float16*)(ws + off); off += (size_t)25 * 128 * 64 * 2;
    _Float16* w1h   = (_Float16*)(ws + off); off += (size_t)9 * 128 * 64 * 2;

    // states h0,c0,h1,c1 are contiguous -> single zero pass each launch
    zero_states_kernel<<<1024, 256, 0, stream>>>(h0, 4 * BB * DD);
    const int nprep = 128 * 64 * 25 + 128 * 64 * 9;
    prep_weights_kernel<<<(nprep + 255) / 256, 256, 0, stream>>>(W0, W1, w0h, w1h);

    for (int t = 0; t < TT; ++t) {
        convlstm_cell_kernel<5, 2, false><<<BB, 256, 0, stream>>>(
            x + (size_t)t * BB * DD, h0, c0, w0h, b0,
            buf16 + (size_t)t * BB * DD, nullptr);
        attention_kernel<<<BB, 256, 0, stream>>>(buf16, att, t);
        convlstm_cell_kernel<3, 1, true><<<BB, 256, 0, stream>>>(
            att, h1, c1, w1h, b1,
            nullptr, out + (size_t)t * BB * DD);
    }
}
